// Encoder_47940424958705
// MI455X (gfx1250) — matrix operand order repaired and checked by reading, not compiled, not run on hardware
//
#include <hip/hip_runtime.h>
#include <math.h>

#define NNODES 50000
#define NEDGES 400000
#define HC     256   // HEADS*HID
#define HIDC   64
#define HEADS  4

typedef __attribute__((ext_vector_type(16))) _Float16 v16h;
typedef __attribute__((ext_vector_type(8)))  float    v8f;

__device__ inline void atomicMaxF32(float* a, float v) {
    int* ai = (int*)a;
    int old = *ai;
    while (__int_as_float(old) < v) {
        int assumed = old;
        old = atomicCAS(ai, assumed, __float_as_int(v));
        if (old == assumed) break;
    }
}

// h[n][c] = bias[c]; amax = -inf; denom = 0
__global__ void k_init(float* __restrict__ hout, const float* __restrict__ bias,
                       float* __restrict__ amax, float* __restrict__ denom) {
    int tid = blockIdx.x * blockDim.x + threadIdx.x;
    if (tid < NNODES * HIDC) hout[tid] = bias[tid & (HIDC - 1)];
    if (tid < NNODES * HEADS) { amax[tid] = -__builtin_inff(); denom[tid] = 0.f; }
}

// Dual GEMM: XL = X @ Wl^T, XR = X @ Wr^T.  X:[M,K] f32, Wl/Wr:[256,K] f32.
// One block (256 thr = 8 waves) owns one 16-row block and ALL 32 output tiles
// (16 col tiles x 2 matrices). The 16xK A tile (contiguous rows of X) is staged
// once into LDS as f16; each wave computes 4 independent 16x16 WMMA tiles.
__global__ void k_gemm_dual_wmma(const float* __restrict__ X,
                                 const float* __restrict__ Wl,
                                 const float* __restrict__ Wr,
                                 float* __restrict__ XL,
                                 float* __restrict__ XR,
                                 int K) {
    __shared__ _Float16 sA[16 * 128];          // max K = 128 -> 4 KB

    const int tid     = threadIdx.x;
    const int rowBase = blockIdx.x * 16;

    // Stage A tile: rows rowBase..rowBase+15 are contiguous 16*K floats.
    const float* Xt = X + (size_t)rowBase * K;
    for (int idx = tid; idx < 16 * K; idx += 256)
        sA[idx] = (_Float16)Xt[idx];
    __syncthreads();

    const int w    = tid >> 5;                 // wave 0..7
    const int lane = tid & 31;
    const int g    = lane >> 4;                // lane group 0/1
    const int ln   = lane & 15;

    const float* Wm  = (w & 4) ? Wr : Wl;      // waves 0-3 -> XL, 4-7 -> XR
    float*       OUT = (w & 4) ? XR : XL;
    const int    ct0 = (w & 3) * 4;            // this wave's 4 column tiles

    v8f acc[4] = {};
    for (int kb = 0; kb < K; kb += 32) {
        // A 16x32 f16 fragment: lane group g holds K in [16g,16g+16)
        const _Float16* ap = sA + ln * K + kb + 16 * g;
        v16h a;
        #pragma unroll
        for (int j = 0; j < 8; ++j) { a[j] = ap[j]; a[8 + j] = ap[8 + j]; }

        #pragma unroll
        for (int t = 0; t < 4; ++t) {
            // B 32x16 f16 fragment: lane group g holds K in [16g,16g+16) of col ln
            const int wrow = (ct0 + t) * 16 + ln;
            const float* bp = Wm + (size_t)wrow * K + kb + 16 * g;
            v16h b;
            #pragma unroll
            for (int j = 0; j < 16; ++j) b[j] = (_Float16)bp[j];
            acc[t] = __builtin_amdgcn_wmma_f32_16x16x32_f16(false, a, false, b,
                                                            (short)0, acc[t],
                                                            false, false);
        }
    }

    // C/D layout: VGPR r, lane -> (row r+8g, col ln)
    #pragma unroll
    for (int t = 0; t < 4; ++t) {
        float* o = OUT + (size_t)rowBase * HC + (ct0 + t) * 16 + ln;
        #pragma unroll
        for (int r = 0; r < 8; ++r)
            o[(size_t)(r + 8 * g) * HC] = acc[t][r];
    }
}

// Wave-per-edge: alpha_raw[e][h] = sum_c att[h][c]*lrelu(xl[src]+xr[dst]+edge_feat)
__global__ void k_edge_alpha(const float* __restrict__ XL, const float* __restrict__ XR,
                             const float* __restrict__ eattr, const int* __restrict__ ei,
                             const float* __restrict__ We, const float* __restrict__ att,
                             float* __restrict__ alpha, float* __restrict__ amax) {
    __shared__ float sWe[HC * 3];
    __shared__ float sAtt[HC];
    for (int i = threadIdx.x; i < HC * 3; i += blockDim.x) sWe[i]  = We[i];
    for (int i = threadIdx.x; i < HC;     i += blockDim.x) sAtt[i] = att[i];
    __syncthreads();

    const int wave = threadIdx.x >> 5;
    const int lane = threadIdx.x & 31;
    const int e    = blockIdx.x * 8 + wave;
    const int src  = ei[e];
    const int dst  = ei[NEDGES + e];
    const float ea0 = eattr[e * 3 + 0];
    const float ea1 = eattr[e * 3 + 1];
    const float ea2 = eattr[e * 3 + 2];

    const int c0 = lane * 8;                          // 8 channels per lane
    const float4* xl4 = (const float4*)(XL + (size_t)src * HC + c0);
    const float4* xr4 = (const float4*)(XR + (size_t)dst * HC + c0);
    float4 a0 = xl4[0], a1 = xl4[1], b0 = xr4[0], b1 = xr4[1];
    float xl[8] = {a0.x, a0.y, a0.z, a0.w, a1.x, a1.y, a1.z, a1.w};
    float xr[8] = {b0.x, b0.y, b0.z, b0.w, b1.x, b1.y, b1.z, b1.w};

    float s = 0.f;
    #pragma unroll
    for (int j = 0; j < 8; ++j) {
        int c = c0 + j;
        float z = xl[j] + xr[j]
                + ea0 * sWe[c * 3] + ea1 * sWe[c * 3 + 1] + ea2 * sWe[c * 3 + 2];
        z = (z > 0.f) ? z : 0.2f * z;
        s += z * sAtt[c];                              // att[h][c%64] == sAtt[c]
    }
    // reduce within 8-lane head group (lanes 8h..8h+7)
    s += __shfl_xor(s, 1);
    s += __shfl_xor(s, 2);
    s += __shfl_xor(s, 4);
    if ((lane & 7) == 0) {
        int h = lane >> 3;
        alpha[(size_t)e * HEADS + h] = s;
        atomicMaxF32(&amax[(size_t)dst * HEADS + h], s);
    }
}

__global__ void k_edge_exp(float* __restrict__ alpha, const float* __restrict__ amax,
                           float* __restrict__ denom, const int* __restrict__ ei) {
    int idx = blockIdx.x * blockDim.x + threadIdx.x;
    if (idx >= NEDGES * HEADS) return;
    int e = idx >> 2, h = idx & 3;
    int dst = ei[NEDGES + e];
    float a = __expf(alpha[idx] - amax[(size_t)dst * HEADS + h]);
    alpha[idx] = a;
    atomicAdd(&denom[(size_t)dst * HEADS + h], a);
}

// out[dst][c] += (1/H) * sum_h alpha_f[e][h] * xl[src][h][c]
__global__ void k_edge_scatter(const float* __restrict__ XL, const int* __restrict__ ei,
                               const float* __restrict__ alpha, const float* __restrict__ denom,
                               float* __restrict__ out) {
    const int wave = threadIdx.x >> 5;
    const int lane = threadIdx.x & 31;
    const int e    = blockIdx.x * 8 + wave;
    const int src  = ei[e];
    const int dst  = ei[NEDGES + e];

    float af[HEADS];
    #pragma unroll
    for (int h = 0; h < HEADS; ++h)
        af[h] = alpha[(size_t)e * HEADS + h] /
                (denom[(size_t)dst * HEADS + h] + 1e-16f);

    const int c = lane * 2;
    float v0 = 0.f, v1 = 0.f;
    #pragma unroll
    for (int h = 0; h < HEADS; ++h) {
        const float2 x2 = *(const float2*)(XL + (size_t)src * HC + h * HIDC + c);
        v0 += af[h] * x2.x;
        v1 += af[h] * x2.y;
    }
    atomicAdd(&out[(size_t)dst * HIDC + c],     0.25f * v0);
    atomicAdd(&out[(size_t)dst * HIDC + c + 1], 0.25f * v1);
}

__global__ void k_prelu(float* __restrict__ out, const float* __restrict__ w) {
    int tid = blockIdx.x * blockDim.x + threadIdx.x;
    if (tid < NNODES * HIDC) {
        float v = out[tid];
        out[tid] = (v >= 0.f) ? v : w[tid & (HIDC - 1)] * v;
    }
}

extern "C" void kernel_launch(void* const* d_in, const int* in_sizes, int n_in,
                              void* d_out, int out_size, void* d_ws, size_t ws_size,
                              hipStream_t stream) {
    const float* x     = (const float*)d_in[0];
    const int*   ei    = (const int*)  d_in[1];
    const float* eattr = (const float*)d_in[2];
    const float* Wl1   = (const float*)d_in[3];
    const float* Wr1   = (const float*)d_in[4];
    const float* We1   = (const float*)d_in[5];
    const float* att1  = (const float*)d_in[6];
    const float* b1    = (const float*)d_in[7];
    const float* Wl2   = (const float*)d_in[8];
    const float* Wr2   = (const float*)d_in[9];
    const float* We2   = (const float*)d_in[10];
    const float* att2  = (const float*)d_in[11];
    const float* b2    = (const float*)d_in[12];
    const float* pw    = (const float*)d_in[13];
    float* out = (float*)d_out;

    float* ws    = (float*)d_ws;
    float* XL    = ws;                                   // N*256
    float* XR    = XL    + (size_t)NNODES * HC;          // N*256
    float* h1    = XR    + (size_t)NNODES * HC;          // N*64
    float* alpha = h1    + (size_t)NNODES * HIDC;        // E*4
    float* amax  = alpha + (size_t)NEDGES * HEADS;       // N*4
    float* denom = amax  + (size_t)NNODES * HEADS;       // N*4

    const int gemmBlocks = NNODES / 16;                  // 3125 row blocks
    const int initBlocks = (NNODES * HIDC + 255) / 256;
    const int edgeBlocks = NEDGES / 8;                   // wave-per-edge, 8 waves/block
    const int ehBlocks   = (NEDGES * HEADS + 255) / 256;

    // ---- layer 1 ----
    k_init<<<initBlocks, 256, 0, stream>>>(h1, b1, amax, denom);
    k_gemm_dual_wmma<<<gemmBlocks, 256, 0, stream>>>(x, Wl1, Wr1, XL, XR, 128);
    k_edge_alpha<<<edgeBlocks, 256, 0, stream>>>(XL, XR, eattr, ei, We1, att1, alpha, amax);
    k_edge_exp<<<ehBlocks, 256, 0, stream>>>(alpha, amax, denom, ei);
    k_edge_scatter<<<edgeBlocks, 256, 0, stream>>>(XL, ei, alpha, denom, h1);

    // ---- layer 2 ----
    k_gemm_dual_wmma<<<gemmBlocks, 256, 0, stream>>>(h1, Wl2, Wr2, XL, XR, 64);
    k_init<<<initBlocks, 256, 0, stream>>>(out, b2, amax, denom);
    k_edge_alpha<<<edgeBlocks, 256, 0, stream>>>(XL, XR, eattr, ei, We2, att2, alpha, amax);
    k_edge_exp<<<ehBlocks, 256, 0, stream>>>(alpha, amax, denom, ei);
    k_edge_scatter<<<edgeBlocks, 256, 0, stream>>>(XL, ei, alpha, denom, out);

    k_prelu<<<(NNODES * HIDC + 255) / 256, 256, 0, stream>>>(out, pw);
}
